// PeakDetector_10496900071801
// MI455X (gfx1250) — matrix operand order repaired and checked by reading, not compiled, not run on hardware
//
#include <hip/hip_runtime.h>
#include <hip/hip_bf16.h>

typedef float  v4f   __attribute__((ext_vector_type(4)));
typedef float  v8f   __attribute__((ext_vector_type(8)));
typedef __bf16 v4bf  __attribute__((ext_vector_type(4)));
typedef __bf16 v8bf  __attribute__((ext_vector_type(8)));
typedef __bf16 v16bf __attribute__((ext_vector_type(16)));

#define BATCH       2048
#define DIM         8192
#define NUM_PEAKS   64
#define NUM_EXPLOIT 51
#define NUM_EXPLORE 13

// Block = 128 threads (4 waves as 2x2). Block tile 128x128, wave tile 64x64.
#define MB   128
#define NB   128
#define KT2  32
#define LDB  40    // LDS row stride in bf16 (x2B = 80B, multiple of 16B)

// ---------------------------------------------------------------------------
// shared helpers: bf16x3 fragment fetch + 48-WMMA slab compute
// ---------------------------------------------------------------------------
__device__ __forceinline__ void compute_slab(const __bf16* __restrict__ pAh,
                                             const __bf16* __restrict__ pAl,
                                             const __bf16* __restrict__ pBh,
                                             const __bf16* __restrict__ pBl,
                                             int wm, int wn, int frow, int fh,
                                             v8f acc[4][4]) {
    // A frags: 16x32 bf16 layout -> lane gets K octets {8*fh, 8*fh+16}
    v16bf fah[4], fal[4];
#pragma unroll
    for (int mi = 0; mi < 4; ++mi) {
        const __bf16* p = pAh + (wm * 64 + mi * 16 + frow) * LDB + 8 * fh;
        v8bf c0 = *(const v8bf*)p;
        v8bf c1 = *(const v8bf*)(p + 16);
        fah[mi] = __builtin_shufflevector(c0, c1, 0,1,2,3,4,5,6,7,8,9,10,11,12,13,14,15);
        const __bf16* q = pAl + (wm * 64 + mi * 16 + frow) * LDB + 8 * fh;
        v8bf d0 = *(const v8bf*)q;
        v8bf d1 = *(const v8bf*)(q + 16);
        fal[mi] = __builtin_shufflevector(d0, d1, 0,1,2,3,4,5,6,7,8,9,10,11,12,13,14,15);
    }
    // B frags: 32x16 bf16 layout -> lane gets K octets {8*fh, 8*fh+16}
#pragma unroll
    for (int ni = 0; ni < 4; ++ni) {
        const __bf16* p = pBh + (wn * 64 + ni * 16 + frow) * LDB + 8 * fh;
        v8bf c0 = *(const v8bf*)p;
        v8bf c1 = *(const v8bf*)(p + 16);
        v16bf fbh = __builtin_shufflevector(c0, c1, 0,1,2,3,4,5,6,7,8,9,10,11,12,13,14,15);
        const __bf16* q = pBl + (wn * 64 + ni * 16 + frow) * LDB + 8 * fh;
        v8bf d0 = *(const v8bf*)q;
        v8bf d1 = *(const v8bf*)(q + 16);
        v16bf fbl = __builtin_shufflevector(d0, d1, 0,1,2,3,4,5,6,7,8,9,10,11,12,13,14,15);
#pragma unroll
        for (int mi = 0; mi < 4; ++mi) {
            acc[mi][ni] = __builtin_amdgcn_wmma_f32_16x16x32_bf16(
                false, fah[mi], false, fbh, (short)0, acc[mi][ni], false, false);
            acc[mi][ni] = __builtin_amdgcn_wmma_f32_16x16x32_bf16(
                false, fah[mi], false, fbl, (short)0, acc[mi][ni], false, false);
            acc[mi][ni] = __builtin_amdgcn_wmma_f32_16x16x32_bf16(
                false, fal[mi], false, fbh, (short)0, acc[mi][ni], false, false);
        }
    }
}

__device__ __forceinline__ void epilogue_store(float* __restrict__ scores,
                                               const float* __restrict__ bias,
                                               int mBase, int nBase,
                                               int wm, int wn, int lane,
                                               v8f acc[4][4]) {
    const int rhi  = (lane >> 4) * 8;
    const int ncol = lane & 15;
#pragma unroll
    for (int mi = 0; mi < 4; ++mi) {
#pragma unroll
        for (int ni = 0; ni < 4; ++ni) {
            const int col = nBase + wn * 64 + ni * 16 + ncol;
            const float bv = bias[col];
#pragma unroll
            for (int v = 0; v < 8; ++v) {
                const int row = mBase + wm * 64 + mi * 16 + rhi + v;
                scores[(size_t)row * DIM + col] = acc[mi][ni][v] + bv;
            }
        }
    }
}

// ---------------------------------------------------------------------------
// Path 1 (preferred): pre-split f32 -> (hi,lo) bf16 once, then async-LDS GEMM
// ---------------------------------------------------------------------------
__global__ __launch_bounds__(256)
void split_bf16(const float* __restrict__ x, __bf16* __restrict__ hi,
                __bf16* __restrict__ lo, size_t n4) {     // n4 = n/4
    size_t i = ((size_t)blockIdx.x * 256 + threadIdx.x);
    if (i >= n4) return;
    v4f v = *(const v4f*)(x + i * 4);
    v4bf h, l;
#pragma unroll
    for (int j = 0; j < 4; ++j) {
        __bf16 hj = (__bf16)v[j];
        h[j] = hj;
        l[j] = (__bf16)(v[j] - (float)hj);
    }
    *(v4bf*)(hi + i * 4) = h;
    *(v4bf*)(lo + i * 4) = l;
}

// 16B global -> LDS async copy (ASYNCcnt-tracked); inline asm = toolchain-portable
__device__ __forceinline__ void async_ld16(unsigned lds_off, const __bf16* g) {
    asm volatile("global_load_async_to_lds_b128 %0, %1, off"
                 :: "v"(lds_off), "v"((unsigned long long)(size_t)g)
                 : "memory");
}
__device__ __forceinline__ void wait_async0() {
    asm volatile("s_wait_asynccnt 0x0" ::: "memory");
}

__global__ __launch_bounds__(128)
void gemm_scores_bf16x3_async(const __bf16* __restrict__ Ah,
                              const __bf16* __restrict__ Al,
                              const __bf16* __restrict__ Wh,
                              const __bf16* __restrict__ Wl,
                              const float* __restrict__ bias,
                              float* __restrict__ scores) {
    __shared__ __bf16 sAh[2][MB * LDB];
    __shared__ __bf16 sAl[2][MB * LDB];
    __shared__ __bf16 sBh[2][NB * LDB];
    __shared__ __bf16 sBl[2][NB * LDB];

    const int t    = threadIdx.x;
    const int lane = t & 31;
    const int wid  = t >> 5;
    const int wm   = wid >> 1;
    const int wn   = wid & 1;
    const int mBase = blockIdx.y * MB;
    const int nBase = blockIdx.x * NB;

    const __bf16* gAh = Ah + (size_t)(mBase + t) * DIM;
    const __bf16* gAl = Al + (size_t)(mBase + t) * DIM;
    const __bf16* gBh = Wh + (size_t)(nBase + t) * DIM;
    const __bf16* gBl = Wl + (size_t)(nBase + t) * DIM;
    // low 32 bits of a generic pointer to __shared__ = LDS byte offset
    const unsigned oAh = (unsigned)(size_t)(const void*)&sAh[0][t * LDB];
    const unsigned oAl = (unsigned)(size_t)(const void*)&sAl[0][t * LDB];
    const unsigned oBh = (unsigned)(size_t)(const void*)&sBh[0][t * LDB];
    const unsigned oBl = (unsigned)(size_t)(const void*)&sBl[0][t * LDB];
    const unsigned bufStep = (unsigned)(MB * LDB * sizeof(__bf16));

    v8f acc[4][4];
#pragma unroll
    for (int mi = 0; mi < 4; ++mi)
#pragma unroll
        for (int ni = 0; ni < 4; ++ni)
            acc[mi][ni] = (v8f){0.f,0.f,0.f,0.f,0.f,0.f,0.f,0.f};

    const int frow = lane & 15;
    const int fh   = lane >> 4;

    // prologue: async-fill buffer 0 with slab 0 (4 planes x 4 x 16B per thread)
#pragma unroll
    for (int c = 0; c < 4; ++c) {
        async_ld16(oAh + c * 16, gAh + c * 8);
        async_ld16(oAl + c * 16, gAl + c * 8);
        async_ld16(oBh + c * 16, gBh + c * 8);
        async_ld16(oBl + c * 16, gBl + c * 8);
    }
    wait_async0();
    __syncthreads();

    int cur = 0;
    for (int kb = KT2; kb < DIM; kb += KT2) {
        // issue async copies of slab kb into the other buffer
        const unsigned bo = (unsigned)(cur ^ 1) * bufStep;
#pragma unroll
        for (int c = 0; c < 4; ++c) {
            async_ld16(oAh + bo + c * 16, gAh + kb + c * 8);
            async_ld16(oAl + bo + c * 16, gAl + kb + c * 8);
            async_ld16(oBh + bo + c * 16, gBh + kb + c * 8);
            async_ld16(oBl + bo + c * 16, gBl + kb + c * 8);
        }
        // overlap: 48 WMMAs on current buffer while the DMA fills the other
        compute_slab(sAh[cur], sAl[cur], sBh[cur], sBl[cur],
                     wm, wn, frow, fh, acc);
        wait_async0();
        __syncthreads();
        cur ^= 1;
    }
    compute_slab(sAh[cur], sAl[cur], sBh[cur], sBl[cur], wm, wn, frow, fh, acc);
    epilogue_store(scores, bias, mBase, nBase, wm, wn, lane, acc);
}

// ---------------------------------------------------------------------------
// Path 2 (fallback, small workspace): in-kernel split + register double buffer
// ---------------------------------------------------------------------------
__device__ __forceinline__ void load_split(const float* __restrict__ g,
                                           v8bf rh[4], v8bf rl[4]) {
#pragma unroll
    for (int c = 0; c < 4; ++c) {
        v4f x0 = *(const v4f*)(g + c * 8);
        v4f x1 = *(const v4f*)(g + c * 8 + 4);
        v8bf h, l;
#pragma unroll
        for (int j = 0; j < 4; ++j) {
            __bf16 h0 = (__bf16)x0[j], h1 = (__bf16)x1[j];
            h[j] = h0; h[j + 4] = h1;
            l[j]     = (__bf16)(x0[j] - (float)h0);
            l[j + 4] = (__bf16)(x1[j] - (float)h1);
        }
        rh[c] = h; rl[c] = l;
    }
}

__device__ __forceinline__ void stage(__bf16* __restrict__ dst, const v8bf r[4],
                                      int t) {
#pragma unroll
    for (int c = 0; c < 4; ++c)
        *(v8bf*)(dst + t * LDB + c * 8) = r[c];
}

__global__ __launch_bounds__(128)
void gemm_scores_bf16x3(const float* __restrict__ A,
                        const float* __restrict__ W,
                        const float* __restrict__ bias,
                        float* __restrict__ scores) {
    __shared__ __bf16 sAh[2][MB * LDB];
    __shared__ __bf16 sAl[2][MB * LDB];
    __shared__ __bf16 sBh[2][NB * LDB];
    __shared__ __bf16 sBl[2][NB * LDB];

    const int t    = threadIdx.x;
    const int lane = t & 31;
    const int wid  = t >> 5;
    const int wm   = wid >> 1;
    const int wn   = wid & 1;
    const int mBase = blockIdx.y * MB;
    const int nBase = blockIdx.x * NB;
    const float* gA = A + (size_t)(mBase + t) * DIM;
    const float* gB = W + (size_t)(nBase + t) * DIM;

    v8f acc[4][4];
#pragma unroll
    for (int mi = 0; mi < 4; ++mi)
#pragma unroll
        for (int ni = 0; ni < 4; ++ni)
            acc[mi][ni] = (v8f){0.f,0.f,0.f,0.f,0.f,0.f,0.f,0.f};

    const int frow = lane & 15;
    const int fh   = lane >> 4;

    v8bf rah[4], ral[4], rbh[4], rbl[4];
    load_split(gA, rah, ral);
    load_split(gB, rbh, rbl);
    stage(sAh[0], rah, t); stage(sAl[0], ral, t);
    stage(sBh[0], rbh, t); stage(sBl[0], rbl, t);
    __syncthreads();

    int cur = 0;
    for (int kb = KT2; kb < DIM; kb += KT2) {
        load_split(gA + kb, rah, ral);
        load_split(gB + kb, rbh, rbl);
        compute_slab(sAh[cur], sAl[cur], sBh[cur], sBl[cur],
                     wm, wn, frow, fh, acc);
        const int nxt = cur ^ 1;
        stage(sAh[nxt], rah, t); stage(sAl[nxt], ral, t);
        stage(sBh[nxt], rbh, t); stage(sBl[nxt], rbl, t);
        __syncthreads();
        cur = nxt;
    }
    compute_slab(sAh[cur], sAl[cur], sBh[cur], sBl[cur], wm, wn, frow, fh, acc);
    epilogue_store(scores, bias, mBase, nBase, wm, wn, lane, acc);
}

// ---------------- Selection: per-row top-51 + 13 masked-random + gather -----
__device__ __forceinline__ float u01_hash(unsigned x) {
    x ^= x >> 16; x *= 0x7feb352du;
    x ^= x >> 15; x *= 0x846ca68bu;
    x ^= x >> 16;
    return (float)(x >> 8) * (1.0f / 16777216.0f);   // [0,1)
}

__global__ __launch_bounds__(256)
void topk_select(const float* __restrict__ scores,
                 const float* __restrict__ field,
                 float* __restrict__ out) {
    __shared__ float s[DIM];
    __shared__ float rv[256];
    __shared__ int   ri[256];
    __shared__ int   sel[NUM_PEAKS];

    const int row = blockIdx.x;
    const int t   = threadIdx.x;
    const float NEG = -3.402823466e38f;

    for (int i = t; i < DIM; i += 256)
        s[i] = scores[(size_t)row * DIM + i];
    __syncthreads();

    for (int kk = 0; kk < NUM_PEAKS; ++kk) {
        if (kk == NUM_EXPLOIT) {
            __syncthreads();
            for (int i = t; i < DIM; i += 256)
                s[i] = u01_hash((unsigned)(row * DIM + i) + 0x9E3779B9u);
            __syncthreads();
            if (t < NUM_EXPLOIT) s[sel[t]] = -1.0f;
            __syncthreads();
        }
        float bv = NEG; int bi = DIM;
        for (int i = t; i < DIM; i += 256) {
            const float v = s[i];
            if (v > bv || (v == bv && i < bi)) { bv = v; bi = i; }
        }
        rv[t] = bv; ri[t] = bi;
        __syncthreads();
        for (int off = 128; off > 0; off >>= 1) {
            if (t < off) {
                const float ov = rv[t + off]; const int oi = ri[t + off];
                if (ov > rv[t] || (ov == rv[t] && oi < ri[t])) { rv[t] = ov; ri[t] = oi; }
            }
            __syncthreads();
        }
        if (t == 0) { sel[kk] = ri[0]; s[ri[0]] = (kk < NUM_EXPLOIT) ? NEG : -1.0f; }
        __syncthreads();
    }

    if (t < NUM_PEAKS)
        out[(size_t)row * NUM_PEAKS + t] = field[(size_t)row * DIM + sel[t]];
}

// ---------------------------------------------------------------------------
extern "C" void kernel_launch(void* const* d_in, const int* in_sizes, int n_in,
                              void* d_out, int out_size, void* d_ws, size_t ws_size,
                              hipStream_t stream) {
    const float* field = (const float*)d_in[0];   // [2048*8192]
    const float* W     = (const float*)d_in[1];   // [8192*8192]
    const float* bias  = (const float*)d_in[2];   // [8192]
    float* out    = (float*)d_out;                // [2048*64]
    float* scores = (float*)d_ws;                 // 64 MB @ offset 0

    const size_t nA = (size_t)BATCH * DIM;        // 16.7M
    const size_t nW = (size_t)DIM * DIM;          // 67.1M
    const size_t offAh = nA * 4;                  //  64 MB
    const size_t offAl = offAh + nA * 2;          //  96 MB
    const size_t offWh = offAl + nA * 2;          // 128 MB
    const size_t offWl = offWh + nW * 2;          // 256 MB
    const size_t need  = offWl + nW * 2;          // 384 MB

    dim3 g1(DIM / NB, BATCH / MB);                // 64 x 16 blocks

    if (ws_size >= need) {
        __bf16* Ah = (__bf16*)((char*)d_ws + offAh);
        __bf16* Al = (__bf16*)((char*)d_ws + offAl);
        __bf16* Wh = (__bf16*)((char*)d_ws + offWh);
        __bf16* Wl = (__bf16*)((char*)d_ws + offWl);
        split_bf16<<<(unsigned)((nA / 4 + 255) / 256), 256, 0, stream>>>(field, Ah, Al, nA / 4);
        split_bf16<<<(unsigned)((nW / 4 + 255) / 256), 256, 0, stream>>>(W, Wh, Wl, nW / 4);
        gemm_scores_bf16x3_async<<<g1, 128, 0, stream>>>(Ah, Al, Wh, Wl, bias, scores);
    } else {
        gemm_scores_bf16x3<<<g1, 128, 0, stream>>>(field, W, bias, scores);
    }
    topk_select<<<BATCH, 256, 0, stream>>>(scores, field, out);
}
